// MiMoV2FlashBlock_79259326480649
// MI455X (gfx1250) — matrix operand order repaired and checked by reading, not compiled, not run on hardware
//
#include <hip/hip_runtime.h>

// ---------------- problem constants ----------------
#define SEQ    2048
#define HIDN   2048
#define NH     32
#define NKVH   8
#define HD     128
#define VDIM   128
#define ROT    64
#define QSCALE 0.08838834764831845f   // 128^-0.5
#define NEGINF (-1.0e9f)

typedef __bf16 bf16_t;
typedef __attribute__((ext_vector_type(16))) __bf16 v16bf;
typedef __attribute__((ext_vector_type(8)))  __bf16 v8bf;
typedef __attribute__((ext_vector_type(8)))  float  v8f;

// ---------------- WMMA fragment helpers (CDNA5 16x16x32 bf16) ----------------
// A (16x32, MxK): lane m = lane&15 holds row m; lanes 0-15 hold K 0..15,
// lanes 16-31 hold K 16..31 (contiguous, matching load_b_frag).
__device__ __forceinline__ v16bf load_a_frag(const bf16_t* base, int ld, int lane) {
    int m  = lane & 15;
    int kh = lane >> 4;                 // 0 or 1
    return *(const v16bf*)(base + (size_t)m * ld + kh * 16);
}
// B (32x16, KxN) supplied as B^T row-major (N rows x K cols): lane n = lane&15
// holds column n, K = (lane>>4)*16 .. +15 contiguous.
__device__ __forceinline__ v16bf load_b_frag(const bf16_t* base, int ld, int lane) {
    int n  = lane & 15;
    int kh = lane >> 4;
    return *(const v16bf*)(base + (size_t)n * ld + kh * 16);
}
__device__ __forceinline__ v8f wmma_bf16(v16bf a, v16bf b, v8f c) {
    return __builtin_amdgcn_wmma_f32_16x16x32_bf16(
        false, a, false, b, (short)0, c, false, false);
}

// ---- CDNA5 async global->LDS DMA (ASYNCcnt-tracked; ISA 10.x / 08_async_tensor)
// LDS dest operand is the 32-bit LDS byte offset == low 32 bits of the flat
// pointer to a __shared__ object (ISA 10.2).
__device__ __forceinline__ void async_b128_to_lds(void* lds_ptr, const void* gptr) {
    unsigned int       ldsoff = (unsigned int)(unsigned long long)(uintptr_t)lds_ptr;
    unsigned long long gaddr  = (unsigned long long)(uintptr_t)gptr;
    asm volatile("global_load_async_to_lds_b128 %0, %1, off"
                 :: "v"(ldsoff), "v"(gaddr) : "memory");
}
__device__ __forceinline__ void wait_async0() {
    asm volatile("s_wait_asynccnt 0" ::: "memory");
}

__device__ __forceinline__ float redmax16(float x) {
    x = fmaxf(x, __shfl_xor(x, 1, 16));
    x = fmaxf(x, __shfl_xor(x, 2, 16));
    x = fmaxf(x, __shfl_xor(x, 4, 16));
    x = fmaxf(x, __shfl_xor(x, 8, 16));
    return x;
}
__device__ __forceinline__ float redsum16(float x) {
    x += __shfl_xor(x, 1, 16);
    x += __shfl_xor(x, 2, 16);
    x += __shfl_xor(x, 4, 16);
    x += __shfl_xor(x, 8, 16);
    return x;
}

// ---------------- kernel 1: f32 -> bf16 cast ----------------
__global__ void cast_f32_bf16(const float* __restrict__ src, bf16_t* __restrict__ dst,
                              size_t n) {
    size_t stride = (size_t)gridDim.x * blockDim.x;
    for (size_t i = (size_t)blockIdx.x * blockDim.x + threadIdx.x; i < n; i += stride)
        dst[i] = (bf16_t)src[i];
}

// ---------------- kernel 2: GEMM  C[M,N] = A[M,K] * W[N,K]^T  (bf16 in, f32 out)
// Block tile 128x128, K-step 32, 256 threads = 8 waves in a 4(M) x 2(N) grid.
// Double-buffered LDS fed by global_load_async_to_lds_b128: one barrier per
// k-step; tile t+1 DMA overlaps the 8 WMMAs of tile t.
__global__ __launch_bounds__(256)
void gemm_bf16_nt(const bf16_t* __restrict__ A, const bf16_t* __restrict__ W,
                  float* __restrict__ C, int M, int N, int K) {
    __shared__ alignas(64) bf16_t sA[2][128 * 32];
    __shared__ alignas(64) bf16_t sB[2][128 * 32];
    const int tid  = threadIdx.x;
    const int lane = tid & 31, wid = tid >> 5;
    const int wm = wid & 3, wn = wid >> 2;
    const int bm = blockIdx.y * 128, bn = blockIdx.x * 128;

    // per-thread staging coords: 512 chunks of 8 bf16 per 128x32 tile
    auto stage = [&](int buf, int k0) {
#pragma unroll
        for (int i = 0; i < 2; ++i) {
            int c = tid + i * 256;
            int row = c >> 2, col8 = (c & 3) * 8;
            async_b128_to_lds(&sA[buf][row * 32 + col8],
                              &A[(size_t)(bm + row) * K + k0 + col8]);
            async_b128_to_lds(&sB[buf][row * 32 + col8],
                              &W[(size_t)(bn + row) * K + k0 + col8]);
        }
    };

    v8f acc[2][4] = {};
    const int T = K / 32;
    stage(0, 0);
    for (int t = 0; t < T; ++t) {
        const int cur = t & 1, k0 = t * 32;
        wait_async0();          // my chunks of tile t are in LDS
        __syncthreads();        // all waves' chunks visible; other buffer free
        if (t + 1 < T) stage(cur ^ 1, k0 + 32);   // DMA next tile, overlap WMMA
        if (k0 + 64 < K) {      // L2 hint for tile t+2
            __builtin_prefetch(&A[(size_t)(bm + (tid >> 1)) * K + k0 + 64], 0, 3);
            __builtin_prefetch(&W[(size_t)(bn + (tid >> 1)) * K + k0 + 64], 0, 3);
        }
        v16bf af[2], bfv[4];
#pragma unroll
        for (int mt = 0; mt < 2; ++mt)
            af[mt] = load_a_frag(&sA[cur][(wm * 32 + mt * 16) * 32], 32, lane);
#pragma unroll
        for (int nt = 0; nt < 4; ++nt)
            bfv[nt] = load_b_frag(&sB[cur][(wn * 64 + nt * 16) * 32], 32, lane);
#pragma unroll
        for (int mt = 0; mt < 2; ++mt)
#pragma unroll
            for (int nt = 0; nt < 4; ++nt)
                acc[mt][nt] = wmma_bf16(af[mt], bfv[nt], acc[mt][nt]);
    }
    // C/D layout: VGPR r -> M = r + 8*(lane>=16); N = lane&15
    const int n = lane & 15, rb = (lane >> 4) * 8;
#pragma unroll
    for (int mt = 0; mt < 2; ++mt)
#pragma unroll
        for (int nt = 0; nt < 4; ++nt)
#pragma unroll
            for (int r = 0; r < 8; ++r) {
                size_t row = (size_t)bm + wm * 32 + mt * 16 + rb + r;
                size_t col = (size_t)bn + wn * 64 + nt * 16 + n;
                C[row * N + col] = acc[mt][nt][r];
            }
}

// ---------------- kernel 3: RoPE + head-major reorg ----------------
__global__ void rope_reorg_q(const float* __restrict__ Qf, const float* __restrict__ cosb,
                             const float* __restrict__ sinb, bf16_t* __restrict__ Qh) {
    size_t i = (size_t)blockIdx.x * blockDim.x + threadIdx.x;   // over [S][H][D]
    if (i >= (size_t)SEQ * NH * HD) return;
    int d = (int)(i % HD);
    int h = (int)((i / HD) % NH);
    int s = (int)(i / ((size_t)HD * NH));
    float x = Qf[i], y;
    if (d < ROT / 2)       y = x * cosb[s * ROT + d] - Qf[i + ROT / 2] * sinb[s * ROT + d];
    else if (d < ROT)      y = x * cosb[s * ROT + d] + Qf[i - ROT / 2] * sinb[s * ROT + d];
    else                   y = x;
    Qh[((size_t)h * SEQ + s) * HD + d] = (bf16_t)(y * QSCALE);  // fold in D^-0.5
}
__global__ void rope_reorg_k(const float* __restrict__ Kf, const float* __restrict__ cosb,
                             const float* __restrict__ sinb, bf16_t* __restrict__ Kh) {
    size_t i = (size_t)blockIdx.x * blockDim.x + threadIdx.x;   // over [S][KVH][D]
    if (i >= (size_t)SEQ * NKVH * HD) return;
    int d  = (int)(i % HD);
    int kh = (int)((i / HD) % NKVH);
    int s  = (int)(i / ((size_t)HD * NKVH));
    float x = Kf[i], y;
    if (d < ROT / 2)       y = x * cosb[s * ROT + d] - Kf[i + ROT / 2] * sinb[s * ROT + d];
    else if (d < ROT)      y = x * cosb[s * ROT + d] + Kf[i - ROT / 2] * sinb[s * ROT + d];
    else                   y = x;
    Kh[((size_t)kh * SEQ + s) * HD + d] = (bf16_t)y;
}
__global__ void reorg_v_t(const float* __restrict__ Vf, bf16_t* __restrict__ Vt) {
    size_t i = (size_t)blockIdx.x * blockDim.x + threadIdx.x;   // over [S][KVH][VD]
    if (i >= (size_t)SEQ * NKVH * VDIM) return;
    int d  = (int)(i % VDIM);
    int kh = (int)((i / VDIM) % NKVH);
    int s  = (int)(i / ((size_t)VDIM * NKVH));
    Vt[((size_t)kh * VDIM + d) * SEQ + s] = (bf16_t)Vf[i];      // transpose: [kvh][vd][s]
}

// ---------------- kernel 4: causal flash attention with sink ----------------
// grid = (S/64, H); 128 threads = 4 waves; each wave owns 16 query rows and
// streams 32-key tiles: scores via 8 WMMAs, P*V via 8 WMMAs per tile.
// Sink column: init m = sink_bias[h], l = exp(sink-m) = 1, never emit a column.
__global__ __launch_bounds__(128)
void attn_flash(const bf16_t* __restrict__ Qh, const bf16_t* __restrict__ Kh,
                const bf16_t* __restrict__ Vt, const float* __restrict__ sink_bias,
                bf16_t* __restrict__ Ctx) {
    __shared__ alignas(64) bf16_t sP[4][16 * 32];   // per-wave P-tile restage
    const int lane = threadIdx.x & 31, wid = threadIdx.x >> 5;
    const int h = blockIdx.y, kvh = h >> 2;
    const int q0 = blockIdx.x * 64 + wid * 16;
    const int col_in = lane & 15, half = lane >> 4;

    v16bf aq[4];
#pragma unroll
    for (int kk = 0; kk < 4; ++kk)
        aq[kk] = load_a_frag(Qh + ((size_t)h * SEQ + q0) * HD + kk * 32, HD, lane);

    float m_r[8], l_r[8], corr[8];
    v8f acc[8] = {};
    const float sink = sink_bias[h];
#pragma unroll
    for (int r = 0; r < 8; ++r) { m_r[r] = sink; l_r[r] = 1.0f; }

    const int kend = blockIdx.x * 64 + 63;          // block-uniform causal bound
    for (int key0 = 0; key0 <= kend; key0 += 32) {
        v8f sc[2];
#pragma unroll
        for (int sub = 0; sub < 2; ++sub) {
            v8f s = {};
#pragma unroll
            for (int kk = 0; kk < 4; ++kk) {
                v16bf bk = load_b_frag(
                    Kh + ((size_t)kvh * SEQ + key0 + sub * 16) * HD + kk * 32, HD, lane);
                s = wmma_bf16(aq[kk], bk, s);
            }
            sc[sub] = s;
        }
        float pnew[2][8];
#pragma unroll
        for (int r = 0; r < 8; ++r) {
            int qrow = q0 + half * 8 + r;
            float s0 = (key0 + col_in      <= qrow) ? sc[0][r] : NEGINF;
            float s1 = (key0 + 16 + col_in <= qrow) ? sc[1][r] : NEGINF;
            float mx = redmax16(fmaxf(s0, s1));
            float mn = fmaxf(m_r[r], mx);
            corr[r] = __expf(m_r[r] - mn);
            float e0 = __expf(s0 - mn);
            float e1 = __expf(s1 - mn);
            l_r[r] = l_r[r] * corr[r] + redsum16(e0 + e1);
            m_r[r] = mn;
            pnew[0][r] = e0; pnew[1][r] = e1;
        }
#pragma unroll
        for (int c = 0; c < 8; ++c)
#pragma unroll
            for (int r = 0; r < 8; ++r) acc[c][r] *= corr[r];
        // restage P (C-layout regs) into row-major LDS, reload as A-fragment
#pragma unroll
        for (int sub = 0; sub < 2; ++sub)
#pragma unroll
            for (int r = 0; r < 8; ++r)
                sP[wid][(half * 8 + r) * 32 + sub * 16 + col_in] = (bf16_t)pnew[sub][r];
        __syncthreads();
        v16bf ap = load_a_frag(&sP[wid][0], 32, lane);
#pragma unroll
        for (int c = 0; c < 8; ++c) {
            v16bf bv = load_b_frag(
                Vt + ((size_t)kvh * VDIM + c * 16) * SEQ + key0, SEQ, lane);
            acc[c] = wmma_bf16(ap, bv, acc[c]);
        }
        __syncthreads();
    }
#pragma unroll
    for (int c = 0; c < 8; ++c)
#pragma unroll
        for (int r = 0; r < 8; ++r) {
            size_t row = (size_t)q0 + half * 8 + r;
            Ctx[row * (NH * VDIM) + (size_t)h * VDIM + c * 16 + col_in] =
                (bf16_t)(acc[c][r] / l_r[r]);
        }
}

// ---------------- host-side orchestration ----------------
extern "C" void kernel_launch(void* const* d_in, const int* in_sizes, int n_in,
                              void* d_out, int out_size, void* d_ws, size_t ws_size,
                              hipStream_t stream) {
    const float* hs   = (const float*)d_in[0];
    const float* cosb = (const float*)d_in[1];
    const float* sinb = (const float*)d_in[2];
    /* d_in[3] attention_mask: exactly causal NEG, implemented analytically */
    const float* Wq   = (const float*)d_in[4];
    const float* Wk   = (const float*)d_in[5];
    const float* Wv   = (const float*)d_in[6];
    const float* Wo   = (const float*)d_in[7];
    const float* sink = (const float*)d_in[8];
    float* out = (float*)d_out;

    // workspace layout (bytes); Ctx reuses the dead Qf region -> 120 MB total
    const size_t MB = 1024ull * 1024ull;
    char* w = (char*)d_ws;
    bf16_t* Xb  = (bf16_t*)(w + 0);          //  8 MB  [S][HID]
    bf16_t* Wqb = (bf16_t*)(w + 8  * MB);    // 16 MB  [H*D][HID]
    bf16_t* Wkb = (bf16_t*)(w + 24 * MB);    //  4 MB
    bf16_t* Wvb = (bf16_t*)(w + 28 * MB);    //  4 MB
    bf16_t* Wob = (bf16_t*)(w + 32 * MB);    // 16 MB  [HID][H*VD]
    float*  Qf  = (float*) (w + 48 * MB);    // 32 MB  [S][H*D]
    float*  Kf  = (float*) (w + 80 * MB);    //  8 MB
    float*  Vf  = (float*) (w + 88 * MB);    //  8 MB
    bf16_t* QhB = (bf16_t*)(w + 96 * MB);    // 16 MB  [H][S][D]
    bf16_t* KhB = (bf16_t*)(w + 112 * MB);   //  4 MB  [KVH][S][D]
    bf16_t* VtB = (bf16_t*)(w + 116 * MB);   //  4 MB  [KVH][VD][S]
    bf16_t* Ctx = (bf16_t*)(w + 48 * MB);    // 16 MB  [S][H*VD] (reuses Qf)

    // 1) casts
    cast_f32_bf16<<<2048, 256, 0, stream>>>(hs, Xb,  (size_t)SEQ * HIDN);
    cast_f32_bf16<<<2048, 256, 0, stream>>>(Wq, Wqb, (size_t)NH * HD * HIDN);
    cast_f32_bf16<<<2048, 256, 0, stream>>>(Wk, Wkb, (size_t)NKVH * HD * HIDN);
    cast_f32_bf16<<<2048, 256, 0, stream>>>(Wv, Wvb, (size_t)NKVH * VDIM * HIDN);
    cast_f32_bf16<<<2048, 256, 0, stream>>>(Wo, Wob, (size_t)HIDN * NH * VDIM);

    // 2) QKV projections (C = X * W^T)
    gemm_bf16_nt<<<dim3((NH * HD) / 128, SEQ / 128), 256, 0, stream>>>(
        Xb, Wqb, Qf, SEQ, NH * HD, HIDN);
    gemm_bf16_nt<<<dim3((NKVH * HD) / 128, SEQ / 128), 256, 0, stream>>>(
        Xb, Wkb, Kf, SEQ, NKVH * HD, HIDN);
    gemm_bf16_nt<<<dim3((NKVH * VDIM) / 128, SEQ / 128), 256, 0, stream>>>(
        Xb, Wvb, Vf, SEQ, NKVH * VDIM, HIDN);

    // 3) RoPE + reorg (+ scale folded into Q, V transposed for WMMA B-frags)
    rope_reorg_q<<<(SEQ * NH * HD) / 256, 256, 0, stream>>>(Qf, cosb, sinb, QhB);
    rope_reorg_k<<<(SEQ * NKVH * HD) / 256, 256, 0, stream>>>(Kf, cosb, sinb, KhB);
    reorg_v_t   <<<(SEQ * NKVH * VDIM) / 256, 256, 0, stream>>>(Vf, VtB);

    // 4) flash attention with sink
    attn_flash<<<dim3(SEQ / 64, NH), 128, 0, stream>>>(QhB, KhB, VtB, sink, Ctx);

    // 5) output projection straight into d_out (f32)
    gemm_bf16_nt<<<dim3(HIDN / 128, SEQ / 128), 256, 0, stream>>>(
        Ctx, Wob, out, SEQ, HIDN, NH * VDIM);
}
